// ODEBlock_53961969107356
// MI455X (gfx1250) — compile-verified, matrix operand order checked by reading, not run on hardware
//
#include <hip/hip_runtime.h>

#define N_NODES 50000
#define HIDDEN  64
#define N_EDGES 800000
#define NUM_STEPS 10
#define FEAT (N_NODES * HIDDEN)

typedef __attribute__((ext_vector_type(16))) _Float16 v16h;
typedef __attribute__((ext_vector_type(8)))  float    v8f;

// ------------------------- precompute kernels -------------------------

__global__ void __launch_bounds__(256) k_set_deg(float* __restrict__ deg) {
  int i = blockIdx.x * blockDim.x + threadIdx.x;
  if (i < N_NODES) deg[i] = 1.0f;  // self-loop contributes 1
}

__global__ void __launch_bounds__(256) k_deg_edges(const int* __restrict__ col,
                                                   float* __restrict__ deg) {
  int e = blockIdx.x * blockDim.x + threadIdx.x;
  if (e < N_EDGES) unsafeAtomicAdd(&deg[col[e]], 1.0f);  // exact integer adds
}

__global__ void __launch_bounds__(256) k_dinv(float* __restrict__ deg) {
  int i = blockIdx.x * blockDim.x + threadIdx.x;
  if (i < N_NODES) deg[i] = rsqrtf(deg[i]);  // deg >= 1 always
}

__global__ void __launch_bounds__(256) k_norm(const int* __restrict__ row,
                                              const int* __restrict__ col,
                                              const float* __restrict__ dinv,
                                              float* __restrict__ norm) {
  int e = blockIdx.x * blockDim.x + threadIdx.x;
  if (e < N_EDGES) norm[e] = dinv[row[e]] * dinv[col[e]];
}

__global__ void __launch_bounds__(256) k_cvt_w(const float* __restrict__ Ws,
                                               const float* __restrict__ Wn,
                                               _Float16* __restrict__ WhS,
                                               _Float16* __restrict__ WhN) {
  int i = blockIdx.x * blockDim.x + threadIdx.x;
  if (i < 2 * HIDDEN * HIDDEN) {
    WhS[i] = (_Float16)Ws[i];
    WhN[i] = (_Float16)Wn[i];
  }
}

// ------------------------- fused dual WMMA GEMM -------------------------
// outS = hin @ Ws^T ; outX = hin @ Wn^T.  Ws/Wn are f16 [64][64] row-major
// (j,k); B fragment element B[k][j] = W[j][k].  One wave owns one 16-row
// tile; block = 4 waves = 64 rows staged once through LDS.  Eight
// independent accumulator chains (2 mats x 4 N-tiles) with the K-chunk loop
// outermost so consecutive WMMAs never have a D->A/B or D->C dependency
// (no hazard v_nops).  781/782 blocks take the branch-free full path.
__global__ void __launch_bounds__(128)
k_gemm_dual(const float* __restrict__ hin,
            const _Float16* __restrict__ Ws,
            const _Float16* __restrict__ Wn,
            float* __restrict__ outS,
            float* __restrict__ outX) {
  __shared__ float tile[64 * 64];
  const int t = threadIdx.x;
  const int row0 = blockIdx.x * 64;
  const bool full = (row0 + 64 <= N_NODES);

  if (full) {
#pragma unroll
    for (int i = 0; i < 8; ++i) {
      int flat = (i * 128 + t) * 4;
      int r = flat >> 6;
      int c = flat & 63;
      *(float4*)(tile + r * 64 + c) =
          *(const float4*)(hin + (size_t)(row0 + r) * HIDDEN + c);
    }
  } else {
#pragma unroll
    for (int i = 0; i < 8; ++i) {
      int flat = (i * 128 + t) * 4;
      int r = flat >> 6;
      int c = flat & 63;
      float4 v = make_float4(0.f, 0.f, 0.f, 0.f);
      if (row0 + r < N_NODES)
        v = *(const float4*)(hin + (size_t)(row0 + r) * HIDDEN + c);
      *(float4*)(tile + r * 64 + c) = v;
    }
  }
  __syncthreads();

  const int wave = t >> 5;
  const int lane = t & 31;
  const int m    = lane & 15;   // M (for A) / N (for B/C)
  const int hi   = lane >> 4;   // half-wave select
  const int wrow = wave * 16;

  // A fragments: 16-bit A 16x32 layout. VGPR i<4: K = 8*hi + 2*i (+1);
  // VGPR i>=4: K = 16 + 8*hi + 2*(i-4) (+1).  Two K-chunks cover K=64.
  v16h afrag[2];
#pragma unroll
  for (int kc = 0; kc < 2; ++kc) {
#pragma unroll
    for (int vi = 0; vi < 8; ++vi) {
      int kb = kc * 32 + ((vi < 4) ? (8 * hi + 2 * vi) : (16 + 8 * hi + 2 * (vi - 4)));
      afrag[kc][2 * vi]     = (_Float16)tile[(wrow + m) * 64 + kb];
      afrag[kc][2 * vi + 1] = (_Float16)tile[(wrow + m) * 64 + kb + 1];
    }
  }

  v8f accv[8];
#pragma unroll
  for (int q = 0; q < 8; ++q)
    accv[q] = (v8f){0.f, 0.f, 0.f, 0.f, 0.f, 0.f, 0.f, 0.f};

#pragma unroll
  for (int kc = 0; kc < 2; ++kc) {
#pragma unroll
    for (int mt = 0; mt < 2; ++mt) {
      const _Float16* W = (mt == 0) ? Ws : Wn;
#pragma unroll
      for (int jt = 0; jt < 4; ++jt) {
        v16h b;  // B 32x16: K along VGPRs (mirrors A), N = m across lanes
#pragma unroll
        for (int vi = 0; vi < 8; ++vi) {
          int kb = kc * 32 + ((vi < 4) ? (8 * hi + 2 * vi) : (16 + 8 * hi + 2 * (vi - 4)));
          b[2 * vi]     = W[(jt * 16 + m) * 64 + kb];
          b[2 * vi + 1] = W[(jt * 16 + m) * 64 + kb + 1];
        }
        accv[mt * 4 + jt] = __builtin_amdgcn_wmma_f32_16x16x32_f16(
            false, afrag[kc], false, b, (short)0, accv[mt * 4 + jt], false, false);
      }
    }
  }

  // C/D layout: VGPR r -> M = r + 8*hi, N = m
#pragma unroll
  for (int mt = 0; mt < 2; ++mt) {
    float* out = (mt == 0) ? outS : outX;
#pragma unroll
    for (int jt = 0; jt < 4; ++jt) {
      v8f cfrag = accv[mt * 4 + jt];
      if (full) {
#pragma unroll
        for (int r = 0; r < 8; ++r)
          out[(size_t)(row0 + wrow + r + 8 * hi) * HIDDEN + jt * 16 + m] = cfrag[r];
      } else {
#pragma unroll
        for (int r = 0; r < 8; ++r) {
          int grow = row0 + wrow + r + 8 * hi;
          if (grow < N_NODES)
            out[(size_t)grow * HIDDEN + jt * 16 + m] = cfrag[r];
        }
      }
    }
  }
}

// ------------------------- edge scatter (L2 atomics) -------------------------
// 16 lanes per edge, float4 per lane: one wave streams two full 256B rows.
__global__ void __launch_bounds__(256)
k_scatter(const int* __restrict__ row, const int* __restrict__ col,
          const float* __restrict__ norm, const float* __restrict__ xn,
          float* __restrict__ agg) {
  long long i = (long long)blockIdx.x * blockDim.x + threadIdx.x;
  if (i >= (long long)N_EDGES * 16) return;
  int e  = (int)(i >> 4);
  int q  = (int)(i & 15);
  int r  = row[e];
  int cn = col[e];
  float nv = norm[e];
  if (q == 0) {  // pull a future gather row toward the caches (global_prefetch_b8)
    int ea = e + 1024;
    if (ea < N_EDGES) __builtin_prefetch(xn + (size_t)row[ea] * HIDDEN, 0, 1);
  }
  float4 v = *(const float4*)(xn + (size_t)r * HIDDEN + q * 4);
  float* dst = agg + (size_t)cn * HIDDEN + q * 4;
  unsafeAtomicAdd(dst + 0, v.x * nv);
  unsafeAtomicAdd(dst + 1, v.y * nv);
  unsafeAtomicAdd(dst + 2, v.z * nv);
  unsafeAtomicAdd(dst + 3, v.w * nv);
}

// ------------------------- combine kernels -------------------------
// layer0: h1 = elu(self + agg + dinv[i]^2 * xn + b0)   (in place into selfb)
__global__ void __launch_bounds__(256)
k_combine_l0(float* selfb, const float* __restrict__ agg,
             const float* __restrict__ xn, const float* __restrict__ dinv,
             const float* __restrict__ bias0) {
  int idx = blockIdx.x * blockDim.x + threadIdx.x;
  if (idx >= FEAT) return;
  int i = idx >> 6;
  int c = idx & 63;
  float d = dinv[i];
  float v = selfb[idx] + agg[idx] + d * d * xn[idx] + bias0[c];
  selfb[idx] = v > 0.f ? v : expm1f(v);
}

// layer1 + RK4 stage update.  self1 aliases htmp (read-before-write per idx).
__global__ void __launch_bounds__(256)
k_combine_l1_rk4(const float* self1, const float* __restrict__ agg,
                 const float* __restrict__ xn, const float* __restrict__ dinv,
                 const float* __restrict__ bias1, const float* __restrict__ tspan,
                 float* hcur, float* htmp, float* acc, int stage) {
  int idx = blockIdx.x * blockDim.x + threadIdx.x;
  if (idx >= FEAT) return;
  float dt = (tspan[1] - tspan[0]) * (1.0f / NUM_STEPS);
  int i = idx >> 6;
  int c = idx & 63;
  float d = dinv[i];
  float k = self1[idx] + agg[idx] + d * d * xn[idx] + bias1[c];
  if (stage == 0) {
    acc[idx] = k;
    htmp[idx] = hcur[idx] + 0.5f * dt * k;
  } else if (stage == 1) {
    acc[idx] += 2.f * k;
    htmp[idx] = hcur[idx] + 0.5f * dt * k;
  } else if (stage == 2) {
    acc[idx] += 2.f * k;
    htmp[idx] = hcur[idx] + dt * k;
  } else {
    hcur[idx] += (dt / 6.f) * (acc[idx] + k);
  }
}

// ------------------------- launch -------------------------

extern "C" void kernel_launch(void* const* d_in, const int* in_sizes, int n_in,
                              void* d_out, int out_size, void* d_ws, size_t ws_size,
                              hipStream_t stream) {
  const float* h       = (const float*)d_in[0];
  const float* W_self  = (const float*)d_in[1];
  const float* W_neigh = (const float*)d_in[2];
  const float* bias    = (const float*)d_in[3];
  const int*   eidx    = (const int*)d_in[4];
  const float* tspan   = (const float*)d_in[5];
  const int* row = eidx;
  const int* col = eidx + N_EDGES;

  float* ws    = (float*)d_ws;
  float* hcur  = ws;
  float* htmp  = ws + 1 * (size_t)FEAT;
  float* acc   = ws + 2 * (size_t)FEAT;
  float* selfb = ws + 3 * (size_t)FEAT;
  float* xnb   = ws + 4 * (size_t)FEAT;
  float* aggb  = ws + 5 * (size_t)FEAT;
  float* dinv  = ws + 6 * (size_t)FEAT;
  float* norm  = dinv + N_NODES;
  _Float16* WhS = (_Float16*)(norm + N_EDGES);
  _Float16* WhN = WhS + 2 * HIDDEN * HIDDEN;

  const int TB = 256;
  const int gbN = (N_NODES + TB - 1) / TB;
  const int gbE = (N_EDGES + TB - 1) / TB;
  const int gbF = (FEAT + TB - 1) / TB;
  const int gbS = (int)(((long long)N_EDGES * 16 + TB - 1) / TB);
  const int gbG = (N_NODES + 63) / 64;
  const int gbW = (2 * HIDDEN * HIDDEN + TB - 1) / TB;

  hipMemcpyAsync(hcur, h, sizeof(float) * FEAT, hipMemcpyDeviceToDevice, stream);
  k_cvt_w<<<gbW, TB, 0, stream>>>(W_self, W_neigh, WhS, WhN);
  k_set_deg<<<gbN, TB, 0, stream>>>(dinv);
  k_deg_edges<<<gbE, TB, 0, stream>>>(col, dinv);
  k_dinv<<<gbN, TB, 0, stream>>>(dinv);
  k_norm<<<gbE, TB, 0, stream>>>(row, col, dinv, norm);

  for (int step = 0; step < NUM_STEPS; ++step) {
    for (int stage = 0; stage < 4; ++stage) {
      const float* hin = (stage == 0) ? hcur : htmp;
      // layer 0
      k_gemm_dual<<<gbG, 128, 0, stream>>>(hin, WhS, WhN, selfb, xnb);
      hipMemsetAsync(aggb, 0, sizeof(float) * FEAT, stream);
      k_scatter<<<gbS, TB, 0, stream>>>(row, col, norm, xnb, aggb);
      k_combine_l0<<<gbF, TB, 0, stream>>>(selfb, aggb, xnb, dinv, bias);
      // layer 1 (self-path output parked in htmp; htmp already consumed)
      k_gemm_dual<<<gbG, 128, 0, stream>>>(selfb, WhS + HIDDEN * HIDDEN,
                                           WhN + HIDDEN * HIDDEN, htmp, xnb);
      hipMemsetAsync(aggb, 0, sizeof(float) * FEAT, stream);
      k_scatter<<<gbS, TB, 0, stream>>>(row, col, norm, xnb, aggb);
      k_combine_l1_rk4<<<gbF, TB, 0, stream>>>(htmp, aggb, xnb, dinv,
                                               bias + HIDDEN, tspan,
                                               hcur, htmp, acc, stage);
    }
  }
  hipMemcpyAsync(d_out, hcur, sizeof(float) * FEAT, hipMemcpyDeviceToDevice, stream);
}
